// CompositionalPINN_59433757442590
// MI455X (gfx1250) — matrix operand order repaired and checked by reading, not compiled, not run on hardware
//
#include <hip/hip_runtime.h>
#include <hip/hip_fp16.h>

typedef __attribute__((ext_vector_type(16))) _Float16 v16h;
typedef __attribute__((ext_vector_type(8)))  _Float16 v8h;
typedef __attribute__((ext_vector_type(8)))  float    v8f;
typedef __attribute__((ext_vector_type(4)))  float    v4f;

// ---------------- workspace (d_ws) layout, in halfs -------------------------
// Weights transposed to [N][K] so B-fragments are contiguous-K reads.
#define WS_W2    0        // 256x256
#define WS_W3    65536    // 256x256
#define WS_W1    131072   // 256x32  (K padded 6->32 with zeros)
#define WS_W4    139264   // 16x256  (N padded 4->16 with zeros)
#define WS_TOTAL 143360   // halfs -> 286720 bytes

// ---------------- LDS layout (bytes) ---------------------------------------
#define OFF_W2 0          // 128 KB
#define OFF_W3 131072     // 128 KB
#define OFF_W1 262144     // 16 KB
#define OFF_W4 278528     // 8 KB
#define OFF_B1 286720     // 1 KB (f32)
#define OFF_B2 287744
#define OFF_B3 288768
#define OFF_B4 289792     // 16 f32 (padded)
#define OFF_H  289856     // + wave*8192 ; 4 waves of 16x256 f16 tiles
#define LDS_BYTES 322624  // fits CDNA5's 320KB-per-workgroup LDS

#define MEMFENCE() __asm__ volatile("" ::: "memory")

__device__ __forceinline__ v8f splat8(float x) {
  v8f r;
#pragma unroll
  for (int i = 0; i < 8; ++i) r[i] = x;
  return r;
}

__device__ __forceinline__ v8f wmma_f16(v16h a, v16h b, v8f c) {
  // v_wmma_f32_16x16x32_f16 : D = A(16x32 f16) x B(32x16 f16) + C(16x16 f32)
  return __builtin_amdgcn_wmma_f32_16x16x32_f16(false, a, false, b, (short)0, c,
                                                false, false);
}

__device__ __forceinline__ float silu(float x) {
  // v_exp (TRANS, co-executes) + v_add + v_rcp + v_mul: no IEEE divide sequence.
  return x * __builtin_amdgcn_rcpf(1.0f + __expf(-x));
}

// Load the 8 A-fragments (K=0..255) for row m from a row-major [16][256] f16
// LDS tile. A-layout: element e of lane L holds K = 16*(L>>4)+e.
__device__ __forceinline__ void load_a_frags(const _Float16* hb, int m, int h,
                                             v16h a[8]) {
#pragma unroll
  for (int kc = 0; kc < 8; ++kc) {
    const _Float16* p = hb + m * 256 + kc * 32 + 16 * h;
    v8h lo = *(const v8h*)(p);        // K = kc*32 + 16h + [0..7]
    v8h hi = *(const v8h*)(p + 8);    // K = kc*32 + 16h + 8 + [0..7]
    v16h av;
#pragma unroll
    for (int i = 0; i < 8; ++i) { av[i] = lo[i]; av[i + 8] = hi[i]; }
    a[kc] = av;
  }
}

// One 256->256 layer: h_out = silu(h_in @ W + b), in/out through LDS tile hb.
// w is [256][256] f16 transposed ([N][K]); bias is 256 f32 in LDS.
// Two N-tiles are processed per iteration as independent WMMA accumulator
// chains so the XDL pipe always has a ready instruction (1 wave/SIMD here).
__device__ __forceinline__ void layer_256(const _Float16* w, const float* bias,
                                          _Float16* hb, int n16, int h) {
  v16h a[8];
  load_a_frags(hb, n16, h, a);   // hoist full-K A fragments before overwriting
  MEMFENCE();
  for (int nt = 0; nt < 16; nt += 2) {
    v8f acc0 = splat8(bias[nt * 16 + n16]);
    v8f acc1 = splat8(bias[(nt + 1) * 16 + n16]);
    const _Float16* wp0 = w + (nt * 16 + n16) * 256 + 16 * h;
    const _Float16* wp1 = wp0 + 16 * 256;
#pragma unroll
    for (int kc = 0; kc < 8; ++kc) {
      // B-fragment: lane = column N, slots = 16 consecutive K per half-wave.
      v16h b0 = *(const v16h*)(wp0 + kc * 32);
      v16h b1 = *(const v16h*)(wp1 + kc * 32);
      acc0 = wmma_f16(a[kc], b0, acc0);
      acc1 = wmma_f16(a[kc], b1, acc1);
    }
#pragma unroll
    for (int v = 0; v < 8; ++v) {   // C layout: M = v + 8h, N = n16
      hb[(v + 8 * h) * 256 + nt * 16 + n16]       = (_Float16)silu(acc0[v]);
      hb[(v + 8 * h) * 256 + (nt + 1) * 16 + n16] = (_Float16)silu(acc1[v]);
    }
  }
  MEMFENCE();
}

// ---------------- prep: f32 weights -> transposed f16 in d_ws ---------------
__global__ void pinn_prep(const float* __restrict__ W1, const float* __restrict__ W2,
                          const float* __restrict__ W3, const float* __restrict__ W4,
                          _Float16* __restrict__ ws) {
  int i = blockIdx.x * blockDim.x + threadIdx.x;
  if (i >= WS_TOTAL) return;
  _Float16 v;
  if (i < WS_W3) {                       // Wt2[n][k] = W2[k][n]
    int j = i - WS_W2, n = j >> 8, k = j & 255;
    v = (_Float16)W2[k * 256 + n];
  } else if (i < WS_W1) {                // Wt3[n][k] = W3[k][n]
    int j = i - WS_W3, n = j >> 8, k = j & 255;
    v = (_Float16)W3[k * 256 + n];
  } else if (i < WS_W4) {                // Wt1[n][k], K padded to 32
    int j = i - WS_W1, n = j >> 5, k = j & 31;
    v = (k < 6) ? (_Float16)W1[k * 256 + n] : (_Float16)0.0f;
  } else {                               // Wt4[n][k], N padded to 16
    int j = i - WS_W4, n = j >> 8, k = j & 255;
    v = (n < 4) ? (_Float16)W4[k * 4 + n] : (_Float16)0.0f;
  }
  ws[i] = v;
}

// ---------------- main kernel ----------------------------------------------
__global__ __launch_bounds__(128) void pinn_main(
    const float* __restrict__ sd, const float* __restrict__ zf,
    const float* __restrict__ b1, const float* __restrict__ b2,
    const float* __restrict__ b3, const float* __restrict__ b4,
    const _Float16* __restrict__ ws, float* __restrict__ out, int B) {
  extern __shared__ __align__(16) char smem[];

  // Stage all weights (f16, 286720 B) into LDS with 16B vector copies.
  {
    const uint4* src = (const uint4*)ws;
    uint4* dst = (uint4*)smem;
    for (int i = threadIdx.x; i < 286720 / 16; i += blockDim.x) dst[i] = src[i];
    for (int t = threadIdx.x; t < 256; t += blockDim.x) {
      ((float*)(smem + OFF_B1))[t] = b1[t];
      ((float*)(smem + OFF_B2))[t] = b2[t];
      ((float*)(smem + OFF_B3))[t] = b3[t];
    }
    if (threadIdx.x < 16)
      ((float*)(smem + OFF_B4))[threadIdx.x] =
          (threadIdx.x < 4) ? b4[threadIdx.x] : 0.0f;
  }
  __syncthreads();

  const int lane = threadIdx.x & 31;
  const int wave = threadIdx.x >> 5;
  const int n16 = lane & 15;     // row of A / column of B / N of C
  const int h   = lane >> 4;     // which half-wave (K/M split)

  const _Float16* w1 = (const _Float16*)(smem + OFF_W1);
  const _Float16* w2 = (const _Float16*)(smem + OFF_W2);
  const _Float16* w3 = (const _Float16*)(smem + OFF_W3);
  const _Float16* w4 = (const _Float16*)(smem + OFF_W4);
  const float* lb1 = (const float*)(smem + OFF_B1);
  const float* lb2 = (const float*)(smem + OFF_B2);
  const float* lb3 = (const float*)(smem + OFF_B3);
  const float* lb4 = (const float*)(smem + OFF_B4);
  _Float16* hb = (_Float16*)(smem + OFF_H + wave * 8192);  // 16x256 f16 tile

  const int NT = (B + 15) / 16;
  for (int t = blockIdx.x * 4 + wave; t < NT; t += gridDim.x * 4) {
    const int rowu = t * 16 + n16;
    const bool valid = rowu < B;
    const int row = valid ? rowu : (B - 1);

    const float* sp = sd + (size_t)row * 6;
    float s0 = sp[0], s1 = sp[1], s2 = sp[2], s3 = sp[3];
    const float qop = sp[4];
    const float dzsub = sp[5] * 0.125f;   // dz_total / N_STEPS
    const float cs = zf[row] * 8.0f;
    const float nf = floorf(cs);
    const float fr = cs - nf;

    for (int step = 0; step < 9; ++step) {
      const bool last = (step == 8);
      const bool run = last ? (fr > 1e-6f) : (nf > (float)step);
      const float dzv = last ? fr * dzsub : dzsub;

      // ---- layer 1: inp(16x6, K-padded to 32) @ W1 -------------------------
      v16h a1;
#pragma unroll
      for (int i = 0; i < 16; ++i) a1[i] = (_Float16)0.0f;
      if (lane < 16) {  // lanes 0-15 hold K=0..7 (only K<6 nonzero)
        a1[0] = (_Float16)s0; a1[1] = (_Float16)s1;
        a1[2] = (_Float16)s2; a1[3] = (_Float16)s3;
        a1[4] = (_Float16)qop; a1[5] = (_Float16)dzv;
      }
      for (int nt = 0; nt < 16; ++nt) {  // 16 independent single-WMMA chains
        v8f acc = splat8(lb1[nt * 16 + n16]);
        v16h b = *(const v16h*)(w1 + (nt * 16 + n16) * 32 + 16 * h);
        acc = wmma_f16(a1, b, acc);
#pragma unroll
        for (int v = 0; v < 8; ++v)
          hb[(v + 8 * h) * 256 + nt * 16 + n16] = (_Float16)silu(acc[v]);
      }
      MEMFENCE();

      // ---- layers 2 & 3: 256 -> 256 ---------------------------------------
      layer_256(w2, lb2, hb, n16, h);
      layer_256(w3, lb3, hb, n16, h);

      // ---- layer 4: 256 -> 16 (4 valid) + residual + mask ------------------
      float o0, o1, o2, o3;
      {
        v16h a[8];
        load_a_frags(hb, n16, h, a);
        MEMFENCE();
        // Split-K: two independent accumulator chains, summed at the end.
        v8f acc0 = splat8(lb4[n16]);
        v8f acc1 = splat8(0.0f);
#pragma unroll
        for (int kc = 0; kc < 4; ++kc) {
          v16h bb0 = *(const v16h*)(w4 + n16 * 256 + (2 * kc) * 32 + 16 * h);
          v16h bb1 = *(const v16h*)(w4 + n16 * 256 + (2 * kc + 1) * 32 + 16 * h);
          acc0 = wmma_f16(a[2 * kc], bb0, acc0);
          acc1 = wmma_f16(a[2 * kc + 1], bb1, acc1);
        }
        float* fs = (float*)hb;  // reuse tile as 16x16 f32 scratch
        MEMFENCE();
#pragma unroll
        for (int v = 0; v < 8; ++v)
          fs[(v + 8 * h) * 16 + n16] = acc0[v] + acc1[v];
        MEMFENCE();
        v4f o = *(const v4f*)(fs + n16 * 16);  // row m=n16, cols 0..3
        o0 = o[0]; o1 = o[1]; o2 = o[2]; o3 = o[3];
      }
      MEMFENCE();
      s0 = run ? s0 + o0 : s0;
      s1 = run ? s1 + o1 : s1;
      s2 = run ? s2 + o2 : s2;
      s3 = run ? s3 + o3 : s3;
    }

    if (lane < 16 && valid) {
      v4f r; r[0] = s0; r[1] = s1; r[2] = s2; r[3] = s3;
      *(v4f*)(out + (size_t)row * 4) = r;
    }
  }
}

extern "C" void kernel_launch(void* const* d_in, const int* in_sizes, int n_in,
                              void* d_out, int out_size, void* d_ws, size_t ws_size,
                              hipStream_t stream) {
  const float* sd = (const float*)d_in[0];   // state_dz (B,6)
  const float* zfr = (const float*)d_in[1];  // z_frac (B,)
  const float* W1 = (const float*)d_in[2];
  const float* b1 = (const float*)d_in[3];
  const float* W2 = (const float*)d_in[4];
  const float* b2 = (const float*)d_in[5];
  const float* W3 = (const float*)d_in[6];
  const float* b3 = (const float*)d_in[7];
  const float* W4 = (const float*)d_in[8];
  const float* b4 = (const float*)d_in[9];
  float* out = (float*)d_out;
  const int B = in_sizes[0] / 6;

  _Float16* ws = (_Float16*)d_ws;

  // 1) transpose + f16-convert weights into workspace (coalesced, tiny).
  pinn_prep<<<(WS_TOTAL + 255) / 256, 256, 0, stream>>>(W1, W2, W3, W4, ws);

  // 2) WMMA MLP. 322KB dynamic LDS -> needs the attribute raised (CDNA5: 320KB/WG).
  (void)hipFuncSetAttribute(reinterpret_cast<const void*>(pinn_main),
                            hipFuncAttributeMaxDynamicSharedMemorySize, LDS_BYTES);
  const int NT = (B + 15) / 16;
  int grid = (NT + 3) / 4;
  if (grid > 512) grid = 512;
  pinn_main<<<grid, 128, LDS_BYTES, stream>>>(sd, zfr, b1, b2, b3, b4, ws, out, B);
}
